// SwitchBiLSTM_46153718563412
// MI455X (gfx1250) — matrix operand order repaired and checked by reading, not compiled, not run on hardware
//
#include <hip/hip_runtime.h>
#include <hip/hip_bf16.h>

// ---------------------------------------------------------------------------
// SwitchBiLSTM for MI455X (gfx1250): bf16 WMMA GEMMs + LDS-resident LSTM scan
// ---------------------------------------------------------------------------

typedef __attribute__((ext_vector_type(16))) __bf16 v16bf;
typedef __attribute__((ext_vector_type(8)))  __bf16 v8bf;
typedef __attribute__((ext_vector_type(8)))  float  v8f;

static constexpr int Bn = 32;     // batch
static constexpr int Tn = 1024;   // time
static constexpr int En = 512;    // embed
static constexpr int Hn = 512;    // hidden
static constexpr int G4 = 2048;   // 4*H gates
static constexpr int TAG = 2;
static constexpr int NTILES = G4 / 16;  // 128 N-tiles over the gate dim

__device__ __forceinline__ v8f wmma_bf16(v16bf a, v16bf b, v8f c) {
  // D = A(16x32 bf16) x B(32x16 bf16) + C(16x16 f32)
  return __builtin_amdgcn_wmma_f32_16x16x32_bf16(
      /*neg_a=*/false, a, /*neg_b=*/false, b,
      /*c_mod=*/(short)0, c, /*reuse_a=*/false, /*reuse_b=*/false);
}

// Combine two contiguous 8xbf16 (16B) chunks into the 16-element per-lane reg.
__device__ __forceinline__ v16bf load2x8(const __bf16* p0, const __bf16* p1) {
  v8bf lo = *reinterpret_cast<const v8bf*>(p0);
  v8bf hi = *reinterpret_cast<const v8bf*>(p1);
  v16bf r;
#pragma unroll
  for (int i = 0; i < 8; ++i) { r[i] = lo[i]; r[8 + i] = hi[i]; }
  return r;
}

// A-matrix (16x32 bf16, MxK): lane&15 = M row; half = lane>>4 selects K blocks:
//   elements 0..7  -> K = k0 + half*8 + 0..7
//   elements 8..15 -> K = k0 + 16 + half*8 + 0..7
__device__ __forceinline__ v16bf load_a_tile(const __bf16* rowk, int half) {
  return load2x8(rowk + half * 8, rowk + 16 + half * 8);
}

// B-matrix (32x16 bf16, KxN): lane&15 = N col; elements 0..7 -> K = k0 + half*8 + e,
//   elements 8..15 -> K = k0 + 16 + half*8 + (e-8)
__device__ __forceinline__ v16bf load_b_tile(const __bf16* colk, int half) {
  return load2x8(colk + half * 8, colk + 16 + half * 8);
}

__device__ __forceinline__ float sigmoidf_(float x) {
  return 1.0f / (1.0f + __expf(-x));
}

__device__ __forceinline__ v8f zero8() {
  v8f z;
#pragma unroll
  for (int i = 0; i < 8; ++i) z[i] = 0.0f;
  return z;
}

// Blocked f32 tile layout for xw (pass-through between proj and lstm, kept in
// WMMA D-layout): element address = (((mtile*128 + ntile)*32 + lane)*8 + r)
__device__ __forceinline__ size_t xw_tile_off(int mtile, int ntile, int lane) {
  return (((size_t)mtile * NTILES + ntile) * 32 + lane) * 8;
}

// ---------------------------------------------------------------------------
// 1) f32 -> bf16 conversion
// ---------------------------------------------------------------------------
__global__ void cvt_bf16_kernel(const float* __restrict__ in,
                                __bf16* __restrict__ out, int n) {
  int i = blockIdx.x * blockDim.x + threadIdx.x;
  int stride = gridDim.x * blockDim.x;
  for (; i < n; i += stride) out[i] = (__bf16)in[i];
}

// ---------------------------------------------------------------------------
// 2) Input projection: xw = xs(row-gathered) @ w_ih^T + bias, stored in
//    blocked WMMA-D tile layout. One wave per (timestep = 2 M-tiles,
//    group of 4 N-tiles): each B tile feeds 2 WMMAs; loads for stage k+1 are
//    software-pipelined against the WMMAs of stage k.
//    reversed=1 applies per-row time reversal of the valid prefix.
// ---------------------------------------------------------------------------
__global__ __launch_bounds__(256) void proj_kernel(
    const __bf16* __restrict__ xs_bf, const int* __restrict__ lengths,
    const __bf16* __restrict__ w_ih_bf, const float* __restrict__ bias,
    float* __restrict__ xw, int reversed) {
  const int lane = threadIdx.x & 31;
  const int wave = threadIdx.x >> 5;
  const int half = lane >> 4;
  const int ln   = lane & 15;

  const int task   = blockIdx.x * 8 + wave;  // 1024 timesteps * 32 N-groups
  const int t      = task >> 5;              // 0..1023
  const int ngroup = task & 31;              // 0..31

  // A rows for this lane for both 16-row M-tiles (A layout: lane&15 = M row)
  const int b0 = ln, b1 = 16 + ln;
  const int len0 = lengths[b0], len1 = lengths[b1];
  const int s0 = reversed ? ((t < len0) ? (len0 - 1 - t) : t) : t;
  const int s1 = reversed ? ((t < len1) ? (len1 - 1 - t) : t) : t;
  const __bf16* arow0 = xs_bf + ((size_t)b0 * Tn + s0) * En;
  const __bf16* arow1 = xs_bf + ((size_t)b1 * Tn + s1) * En;

  v8f acc[2][4];
#pragma unroll
  for (int mt = 0; mt < 2; ++mt)
#pragma unroll
    for (int q = 0; q < 4; ++q) acc[mt][q] = zero8();

  // ---- software-pipelined K loop (depth 1) ----
  v16bf a0c = load_a_tile(arow0, half);
  v16bf a1c = load_a_tile(arow1, half);
  v16bf bc[4];
#pragma unroll
  for (int q = 0; q < 4; ++q) {
    const int j = (ngroup * 4 + q) * 16 + ln;
    bc[q] = load_b_tile(w_ih_bf + (size_t)j * En, half);
  }

#pragma unroll 4
  for (int kk = 0; kk < En; kk += 32) {
    // issue stage k+1 loads (branchless: last iteration re-loads itself)
    const int kn = (kk + 32 < En) ? (kk + 32) : kk;
    v16bf a0n = load_a_tile(arow0 + kn, half);
    v16bf a1n = load_a_tile(arow1 + kn, half);
    v16bf bn[4];
#pragma unroll
    for (int q = 0; q < 4; ++q) {
      const int j = (ngroup * 4 + q) * 16 + ln;
      bn[q] = load_b_tile(w_ih_bf + (size_t)j * En + kn, half);
    }
    // compute stage k while stage k+1 is in flight
#pragma unroll
    for (int q = 0; q < 4; ++q) {
      acc[0][q] = wmma_bf16(a0c, bc[q], acc[0][q]);
      acc[1][q] = wmma_bf16(a1c, bc[q], acc[1][q]);
    }
    a0c = a0n; a1c = a1n;
#pragma unroll
    for (int q = 0; q < 4; ++q) bc[q] = bn[q];
  }

  // store D tiles in blocked layout: per lane 8 contiguous floats (32B)
#pragma unroll
  for (int mt = 0; mt < 2; ++mt)
#pragma unroll
    for (int q = 0; q < 4; ++q) {
      const int ntile = ngroup * 4 + q;
      const int j = ntile * 16 + ln;
      const float bj = bias[j];
      v8f res;
#pragma unroll
      for (int r = 0; r < 8; ++r) res[r] = acc[mt][q][r] + bj;
      *reinterpret_cast<v8f*>(xw + xw_tile_off(t * 2 + mt, ntile, lane)) = res;
    }
}

// ---------------------------------------------------------------------------
// 3) LSTM recurrence. grid.x = 2 (direction), 1024 threads = 32 waves.
//    h (bf16, ping-pong double buffer) and c (f32) live in LDS. Wave w owns
//    H-columns [w*16, w*16+16) and computes i/f/g/o gate tiles for BOTH
//    16-row M-tiles -> the cell update is wave-local. One barrier per step.
//    Latency hiding is cross-wave (8 waves/SIMD resident); VGPR footprint is
//    kept ~110 so all 32 waves fit the WGP without spills.
// ---------------------------------------------------------------------------
static constexpr int HP = Hn + 16;  // bf16 pitch for h rows
static constexpr int CP = Hn + 4;   // f32 pitch for c rows

__global__ __launch_bounds__(1024) void lstm_kernel(
    const float* __restrict__ xw_f, const float* __restrict__ xw_b,
    const __bf16* __restrict__ whh_f, const __bf16* __restrict__ whh_b,
    const int* __restrict__ lengths,
    float* __restrict__ out_f, float* __restrict__ out_b) {
  const int dir = blockIdx.x;
  const float* __restrict__ xw   = dir ? xw_b : xw_f;
  const __bf16* __restrict__ whh = dir ? whh_b : whh_f;
  float* __restrict__ out        = dir ? out_b : out_f;

  extern __shared__ char smem[];
  __bf16* h0 = (__bf16*)smem;                                   // [32][HP]
  __bf16* h1 = h0 + (size_t)Bn * HP;                            // [32][HP]
  float*  c_sh = (float*)(smem + 2 * (size_t)Bn * HP * sizeof(__bf16));

  const int lane = threadIdx.x & 31;
  const int wave = threadIdx.x >> 5;  // coltile 0..31
  const int half = lane >> 4;
  const int ln   = lane & 15;
  const int coltile = wave;
  const int col = coltile * 16 + ln;

  for (int i = threadIdx.x; i < 2 * Bn * HP; i += 1024) h0[i] = (__bf16)0.0f;
  for (int i = threadIdx.x; i < Bn * CP; i += 1024) c_sh[i] = 0.0f;

  int lenb[2][8];
#pragma unroll
  for (int mt = 0; mt < 2; ++mt)
#pragma unroll
    for (int r = 0; r < 8; ++r) lenb[mt][r] = lengths[mt * 16 + half * 8 + r];

  __syncthreads();

  __bf16* h_rd = h0;
  __bf16* h_wr = h1;

  for (int t = 0; t < Tn; ++t) {
    // init accumulators from the blocked xw tiles (vector loads), and
    // prefetch the next timestep's tiles from HBM while we compute
    v8f acc[2][4];
#pragma unroll
    for (int mt = 0; mt < 2; ++mt)
#pragma unroll
      for (int q = 0; q < 4; ++q) {
        const int ntile = q * 32 + coltile;
        acc[mt][q] = *reinterpret_cast<const v8f*>(
            xw + xw_tile_off(t * 2 + mt, ntile, lane));
        if (t + 1 < Tn)
          __builtin_prefetch(xw + xw_tile_off((t + 1) * 2 + mt, ntile, lane),
                             0, 3);
      }

    // g += h @ w_hh^T  (A from LDS, B from L2-resident bf16 weights)
#pragma unroll 2
    for (int kk = 0; kk < Hn; kk += 32) {
      v16bf a0 = load_a_tile(h_rd + (size_t)(0 * 16 + ln) * HP + kk, half);
      v16bf a1 = load_a_tile(h_rd + (size_t)(1 * 16 + ln) * HP + kk, half);
      v16bf bm[4];
#pragma unroll
      for (int q = 0; q < 4; ++q) {
        const int j = q * Hn + col;
        bm[q] = load_b_tile(whh + (size_t)j * Hn + kk, half);
      }
#pragma unroll
      for (int q = 0; q < 4; ++q) {
        acc[0][q] = wmma_bf16(a0, bm[q], acc[0][q]);
        acc[1][q] = wmma_bf16(a1, bm[q], acc[1][q]);
      }
    }

    // LSTM cell update (torch gate order i,f,g,o), state frozen past length.
    // Reads h_rd / writes h_wr -> no barrier needed between phases.
#pragma unroll
    for (int mt = 0; mt < 2; ++mt) {
#pragma unroll
      for (int r = 0; r < 8; ++r) {
        const int b = mt * 16 + half * 8 + r;
        const bool valid = t < lenb[mt][r];
        const float ig = sigmoidf_(acc[mt][0][r]);
        const float fg = sigmoidf_(acc[mt][1][r]);
        const float gg = tanhf(acc[mt][2][r]);
        const float og = sigmoidf_(acc[mt][3][r]);
        const float c_old = c_sh[(size_t)b * CP + col];
        const float c_new = fg * c_old + ig * gg;
        const float h_new = og * tanhf(c_new);
        const float h_old = (float)h_rd[(size_t)b * HP + col];
        c_sh[(size_t)b * CP + col] = valid ? c_new : c_old;
        h_wr[(size_t)b * HP + col] = (__bf16)(valid ? h_new : h_old);
        out[((size_t)t * Bn + b) * Hn + col] = valid ? h_new : 0.0f;
      }
    }

    __syncthreads();  // h_wr complete before it becomes next step's h_rd
    __bf16* tmp = h_rd; h_rd = h_wr; h_wr = tmp;
  }
}

// ---------------------------------------------------------------------------
// 4) Classifier + softmax. One wave per (b,t); lanes split K=1024 (fwd 512 +
//    time-unreversed bwd 512), shfl-xor reduction, softmax over 2 logits.
// ---------------------------------------------------------------------------
__global__ __launch_bounds__(256) void cls_softmax_kernel(
    const float* __restrict__ out_f, const float* __restrict__ out_b,
    const int* __restrict__ lengths,
    const float* __restrict__ w_cls, const float* __restrict__ b_cls,
    float* __restrict__ out) {
  const int lane = threadIdx.x & 31;
  const int wid = (blockIdx.x * blockDim.x + threadIdx.x) >> 5;  // 0..32767
  const int b = wid >> 10;
  const int t = wid & (Tn - 1);
  const int len = lengths[b];
  const bool valid = t < len;
  const int tb = len - 1 - t;  // only used when valid

  float a0 = 0.0f, a1 = 0.0f;
  for (int k = lane; k < Hn; k += 32) {
    const float vf = out_f[((size_t)t * Bn + b) * Hn + k];
    a0 += vf * w_cls[k];
    a1 += vf * w_cls[2 * Hn + k];
    const float vb = valid ? out_b[((size_t)tb * Bn + b) * Hn + k] : 0.0f;
    a0 += vb * w_cls[Hn + k];
    a1 += vb * w_cls[2 * Hn + Hn + k];
  }
#pragma unroll
  for (int off = 16; off > 0; off >>= 1) {
    a0 += __shfl_xor(a0, off, 32);
    a1 += __shfl_xor(a1, off, 32);
  }
  if (lane == 0) {
    const float l0 = a0 + b_cls[0];
    const float l1 = a1 + b_cls[1];
    const float m = fmaxf(l0, l1);
    const float e0 = __expf(l0 - m);
    const float e1 = __expf(l1 - m);
    const float inv = 1.0f / (e0 + e1);
    out[((size_t)b * Tn + t) * TAG + 0] = e0 * inv;
    out[((size_t)b * Tn + t) * TAG + 1] = e1 * inv;
  }
}

// ---------------------------------------------------------------------------
// Launch
// ---------------------------------------------------------------------------
extern "C" void kernel_launch(void* const* d_in, const int* in_sizes, int n_in,
                              void* d_out, int out_size, void* d_ws, size_t ws_size,
                              hipStream_t stream) {
  (void)in_sizes; (void)n_in; (void)out_size; (void)ws_size;
  const float* xs     = (const float*)d_in[0];
  const int* lengths  = (const int*)d_in[1];
  /* d_in[2] = domains (single-domain path, unused) */
  const float* w_ih_f = (const float*)d_in[3];
  const float* w_hh_f = (const float*)d_in[4];
  const float* b_f    = (const float*)d_in[5];
  const float* w_ih_b = (const float*)d_in[6];
  const float* w_hh_b = (const float*)d_in[7];
  const float* b_b    = (const float*)d_in[8];
  const float* w_cls  = (const float*)d_in[9];
  const float* b_cls  = (const float*)d_in[10];
  float* out = (float*)d_out;

  char* ws = (char*)d_ws;
  size_t off = 0;
  auto carve = [&](size_t bytes) -> void* {
    void* p = ws + off;
    off = (off + bytes + 255) & ~(size_t)255;
    return p;
  };
  __bf16* xs_bf   = (__bf16*)carve((size_t)Bn * Tn * En * 2);
  __bf16* wihf_bf = (__bf16*)carve((size_t)G4 * En * 2);
  __bf16* wihb_bf = (__bf16*)carve((size_t)G4 * En * 2);
  __bf16* whhf_bf = (__bf16*)carve((size_t)G4 * Hn * 2);
  __bf16* whhb_bf = (__bf16*)carve((size_t)G4 * Hn * 2);
  float* xw_f  = (float*)carve((size_t)Tn * Bn * G4 * 4);  // blocked tile layout
  float* xw_b  = (float*)carve((size_t)Tn * Bn * G4 * 4);
  float* out_f = (float*)carve((size_t)Tn * Bn * Hn * 4);
  float* out_b = (float*)carve((size_t)Tn * Bn * Hn * 4);

  // 1) convert to bf16
  cvt_bf16_kernel<<<4096, 256, 0, stream>>>(xs, xs_bf, Bn * Tn * En);
  cvt_bf16_kernel<<<1024, 256, 0, stream>>>(w_ih_f, wihf_bf, G4 * En);
  cvt_bf16_kernel<<<1024, 256, 0, stream>>>(w_ih_b, wihb_bf, G4 * En);
  cvt_bf16_kernel<<<1024, 256, 0, stream>>>(w_hh_f, whhf_bf, G4 * Hn);
  cvt_bf16_kernel<<<1024, 256, 0, stream>>>(w_hh_b, whhb_bf, G4 * Hn);

  // 2) input projections (fwd + time-reversed bwd): 32768 wave-tasks each
  proj_kernel<<<4096, 256, 0, stream>>>(xs_bf, lengths, wihf_bf, b_f, xw_f, 0);
  proj_kernel<<<4096, 256, 0, stream>>>(xs_bf, lengths, wihb_bf, b_b, xw_b, 1);

  // 3) recurrence: one workgroup per direction, 32 waves, LDS-resident state
  const size_t smem = 2 * (size_t)Bn * HP * sizeof(__bf16) +
                      (size_t)Bn * CP * sizeof(float);
  lstm_kernel<<<2, 1024, smem, stream>>>(xw_f, xw_b, whhf_bf, whhb_bf, lengths,
                                         out_f, out_b);

  // 4) classifier + softmax: one wave per (b,t)
  cls_softmax_kernel<<<4096, 256, 0, stream>>>(out_f, out_b, lengths, w_cls,
                                               b_cls, out);
}
